// PRFQueryExpander_14542759264481
// MI455X (gfx1250) — matrix operand order repaired and checked by reading, not compiled, not run on hardware
//
#include <hip/hip_runtime.h>
#include <hip/hip_bf16.h>

// ---------------------------------------------------------------------------
// PRF query expander for MI455X (gfx1250, wave32, WMMA).
//
// Roofline: the 131072x300 f32 embedding gather (157 MB) bounds the main
// kernel at ~6.7us @ 23.3 TB/s; the 10 GFLOP scorer GEMM is absorbed by
// v_wmma_f32_16x16x32_f16 (f16 in-register convert, f32 accumulate).
// ---------------------------------------------------------------------------

typedef __attribute__((ext_vector_type(16))) _Float16 v16h;
typedef __attribute__((ext_vector_type(8)))  float    v8f;

#define EMB 300
#define HID 128
#define KSTEPS 10              // ceil(300/32)
#define TERMS_PER_BLOCK 64     // 4 waves x 16 terms
#define SCORE_THREADS 128      // 4 waves

// ---------------------------------------------------------------- zero weights
__global__ void zero_kernel(float* __restrict__ w, int n) {
    int i = blockIdx.x * blockDim.x + threadIdx.x;
    int stride = gridDim.x * blockDim.x;
    for (; i < n; i += stride) w[i] = 0.0f;
}

// ------------------------------------------- query_ctx + constant hidden bias
// c[k] = sb1[k] + sum_j qctx[j] * sW1[(300+j)*128 + k]
__global__ void prep_kernel(const int* __restrict__ qterms, int nq,
                            const float* __restrict__ emb,
                            const float* __restrict__ sW1,
                            const float* __restrict__ sb1,
                            float* __restrict__ qctx,
                            float* __restrict__ cvec) {
    __shared__ float sq[EMB];
    int t = threadIdx.x;                 // 320 threads
    if (t < EMB) {
        float s = 0.0f;
        for (int i = 0; i < nq; ++i)
            s += emb[(size_t)qterms[i] * EMB + t];
        s /= (float)nq;
        sq[t]   = s;
        qctx[t] = s;
    }
    __syncthreads();
    if (t < HID) {
        float s = sb1[t];
        for (int j = 0; j < EMB; ++j)
            s += sq[j] * sW1[(size_t)(EMB + j) * HID + t];
        cvec[t] = s;
    }
}

// ------------------------------------ pack sW1[:300,:] into WMMA B fragments
// B is 32x16 (KxN) f16. Layout (contiguous K order used here, not the ISA table order):
//   lanes 0-15 : N = lane,    K = 0..15  (VGPR j holds K=2j,2j+1)
//   lanes 16-31: N = lane-16, K = 16..31
// Fragment index: ((kstep*8 + ntile)*32 + lane) * 16 halves.
__global__ void pack_kernel(const float* __restrict__ sW1,
                            _Float16* __restrict__ Bpack) {
    int idx = blockIdx.x * blockDim.x + threadIdx.x;
    if (idx >= KSTEPS * 8 * 32) return;
    int lane  = idx & 31;
    int ntile = (idx >> 5) & 7;
    int kstep = idx >> 8;
    int n  = ntile * 16 + (lane & 15);
    int kb = kstep * 32 + ((lane & 16) ? 16 : 0);
    v16h frag;
#pragma unroll
    for (int j = 0; j < 16; ++j) {
        int k = kb + j;
        float v = sW1[(size_t)(k < EMB ? k : 0) * HID + n];
        if (k >= EMB) v = 0.0f;
        frag[j] = (_Float16)v;
    }
    *((v16h*)(Bpack + (size_t)idx * 16)) = frag;
}

// -------------------------------------------------------- main scorer kernel
// Per block: 4 waves x 16 terms. Per wave: D(16x128) = A(16x320_f16) x B.
// 10 k-steps x 8 n-tiles = 80 v_wmma_f32_16x16x32_f16 per wave.
__global__ void __launch_bounds__(SCORE_THREADS)
score_kernel(const int* __restrict__ doc_ids, int N,
             const float* __restrict__ emb,
             const _Float16* __restrict__ Bpack,
             const float* __restrict__ cvec,
             const float* __restrict__ sW2,
             const float* __restrict__ sb2,
             float* __restrict__ weights) {
    __shared__ float hbuf[4][16][HID];   // 32 KB: per-wave 16x128 hidden tile
    __shared__ float s_c[HID];
    __shared__ float s_w2[HID];

    int tid = threadIdx.x;
    if (tid < HID) { s_c[tid] = cvec[tid]; s_w2[tid] = sW2[tid]; }
    __syncthreads();

    int wave = tid >> 5;
    int lane = tid & 31;
    int m    = lane & 15;                       // A row this lane feeds
    int term = blockIdx.x * TERMS_PER_BLOCK + wave * 16 + m;
    int tc   = term < N ? term : N - 1;
    const float* row = emb + (size_t)doc_ids[tc] * EMB;
    __builtin_prefetch(row, 0, 1);              // global_prefetch_b8

    v8f acc[8];
#pragma unroll
    for (int n = 0; n < 8; ++n) acc[n] = (v8f){0,0,0,0,0,0,0,0};

    const v16h* bp = (const v16h*)Bpack;

    for (int ks = 0; ks < KSTEPS; ++ks) {
        // A fragment (contiguous K order, matching Bpack):
        //   lanes 0-15 : halves 0-15 -> K=kb..kb+15, kb=ks*32
        //   lanes 16-31: same with kb=ks*32+16
        v16h a;
        int kb = ks * 32 + ((lane & 16) ? 16 : 0);
#pragma unroll
        for (int j = 0; j < 8; ++j) {
            int k1 = kb + j;
            int k2 = kb + 8 + j;
            float v1 = row[k1 < EMB ? k1 : 0]; if (k1 >= EMB) v1 = 0.0f;
            float v2 = row[k2 < EMB ? k2 : 0]; if (k2 >= EMB) v2 = 0.0f;
            a[j]     = (_Float16)v1;
            a[8 + j] = (_Float16)v2;
        }
#pragma unroll
        for (int n = 0; n < 8; ++n) {
            v16h b = bp[(size_t)(ks * 8 + n) * 32 + lane];
            acc[n] = __builtin_amdgcn_wmma_f32_16x16x32_f16(
                false, a, false, b, (short)0, acc[n], false, false);
        }
    }

    // C/D layout: lanes 0-15 hold rows 0..7 (VGPR r -> M=r), lanes 16-31 rows 8..15.
    int rbase = (lane & 16) ? 8 : 0;
    int col0  = lane & 15;
#pragma unroll
    for (int r = 0; r < 8; ++r) {
#pragma unroll
        for (int n = 0; n < 8; ++n) {
            int col = n * 16 + col0;
            float v = acc[n][r] + s_c[col];
            hbuf[wave][rbase + r][col] = v > 0.0f ? v : 0.0f;   // bias + ReLU
        }
    }
    __syncthreads();

    // score = sigmoid(h . sW2 + sb2); scatter-add into weights[vocab_id]
    if (tid < TERMS_PER_BLOCK) {
        int t = blockIdx.x * TERMS_PER_BLOCK + tid;
        if (t < N) {
            const float* h = hbuf[tid >> 4][tid & 15];
            float s = sb2[0];
#pragma unroll 8
            for (int c = 0; c < HID; ++c) s += h[c] * s_w2[c];
            float score = 1.0f / (1.0f + __expf(-s));
            atomicAdd(&weights[doc_ids[t]], score);
        }
    }
}

// ------------------------------------------------------------------- top-k
__global__ void topk_kernel(float* __restrict__ weights, int vocab,
                            const int* __restrict__ ntp,
                            int* __restrict__ tids, float* __restrict__ tw) {
    __shared__ float sv[256];
    __shared__ int   si[256];
    int nt = *ntp;
    for (int pass = 0; pass < nt; ++pass) {
        float best = -1e30f;
        int   bid  = 0x7fffffff;
        for (int i = threadIdx.x; i < vocab; i += 256) {
            float v = weights[i];
            if (v > best || (v == best && i < bid)) { best = v; bid = i; }
        }
        sv[threadIdx.x] = best; si[threadIdx.x] = bid;
        __syncthreads();
        for (int s = 128; s > 0; s >>= 1) {
            if (threadIdx.x < s) {
                if (sv[threadIdx.x + s] > sv[threadIdx.x] ||
                    (sv[threadIdx.x + s] == sv[threadIdx.x] &&
                     si[threadIdx.x + s] <  si[threadIdx.x])) {
                    sv[threadIdx.x] = sv[threadIdx.x + s];
                    si[threadIdx.x] = si[threadIdx.x + s];
                }
            }
            __syncthreads();
        }
        if (threadIdx.x == 0) {
            tids[pass] = si[0];
            tw[pass]   = sv[0];
            weights[si[0]] = -1e30f;     // exclude from next pass
        }
        __syncthreads();
    }
}

// -------------------------------------------------- expansion MLP + outputs
__global__ void expand_kernel(const float* __restrict__ emb,
                              const float* __restrict__ qctx,
                              const int* __restrict__ tids,
                              const float* __restrict__ tw,
                              const int* __restrict__ ntp,
                              const float* __restrict__ eW1,
                              const float* __restrict__ eb1,
                              const float* __restrict__ eW2,
                              const float* __restrict__ eb2,
                              float* __restrict__ out) {
    __shared__ float feat[3 * EMB];
    __shared__ float hid[HID];
    int nt = *ntp;
    int t = threadIdx.x;                 // 320 threads
    if (t < EMB) {
        float em = 0.0f;
        for (int i = 0; i < nt; ++i)
            em += emb[(size_t)tids[i] * EMB + t];
        em /= (float)nt;
        float q = qctx[t];
        feat[t]           = q;
        feat[EMB + t]     = em;
        feat[2 * EMB + t] = q * em;
    }
    __syncthreads();
    if (t < HID) {
        float s = eb1[t];
        for (int j = 0; j < 3 * EMB; ++j)
            s += feat[j] * eW1[(size_t)j * HID + t];
        hid[t] = s > 0.0f ? s : 0.0f;
    }
    __syncthreads();
    if (t < EMB) {
        float s = eb2[t];
#pragma unroll 8
        for (int k = 0; k < HID; ++k)
            s += hid[k] * eW2[(size_t)k * EMB + t];
        out[t] = s;
    }
    if (t < nt) {
        out[EMB + t]      = (float)tids[t];   // top_ids (output buffer is f32)
        out[EMB + nt + t] = tw[t];            // top_w
    }
}

// ---------------------------------------------------------------------------
extern "C" void kernel_launch(void* const* d_in, const int* in_sizes, int n_in,
                              void* d_out, int out_size, void* d_ws, size_t ws_size,
                              hipStream_t stream) {
    const int*   qterms = (const int*)  d_in[0];
    const int*   docs   = (const int*)  d_in[1];
    const int*   ntp    = (const int*)  d_in[2];
    const float* emb    = (const float*)d_in[3];
    const float* sW1    = (const float*)d_in[4];
    const float* sb1    = (const float*)d_in[5];
    const float* sW2    = (const float*)d_in[6];
    const float* sb2    = (const float*)d_in[7];
    const float* eW1    = (const float*)d_in[8];
    const float* eb1    = (const float*)d_in[9];
    const float* eW2    = (const float*)d_in[10];
    const float* eb2    = (const float*)d_in[11];

    int nq    = in_sizes[0];          // 32 query terms
    int N     = in_sizes[1];          // 131072 flat feedback term ids
    int vocab = in_sizes[3] / EMB;    // 100000

    // Workspace layout (bytes): qctx(320f) | cvec(128f) | Bpack(40960 f16,
    // 32B-aligned at byte 1792) | weights(vocab f) | tids(32i) | tw(32f)
    float*    qctx    = (float*)d_ws;
    float*    cvec    = qctx + 320;
    _Float16* Bpack   = (_Float16*)(cvec + HID);
    float*    weights = (float*)(Bpack + (size_t)KSTEPS * 8 * 32 * 16);
    int*      tids    = (int*)(weights + vocab);
    float*    tw      = (float*)(tids + 32);

    // 1) zero the scatter-add accumulator (top-k clobbers it each call)
    zero_kernel<<<(vocab + 255) / 256, 256, 0, stream>>>(weights, vocab);

    // 2) query context + constant hidden-bias vector
    prep_kernel<<<1, 320, 0, stream>>>(qterms, nq, emb, sW1, sb1, qctx, cvec);

    // 3) pack scorer weights into f16 WMMA B-fragments (80 KB, L2-resident)
    pack_kernel<<<(KSTEPS * 8 * 32 + 255) / 256, 256, 0, stream>>>(sW1, Bpack);

    // 4) WMMA scorer + sigmoid + scatter-add
    int blocks = (N + TERMS_PER_BLOCK - 1) / TERMS_PER_BLOCK;
    score_kernel<<<blocks, SCORE_THREADS, 0, stream>>>(
        docs, N, emb, Bpack, cvec, sW2, sb2, weights);

    // 5) top-n_terms by accumulated weight
    topk_kernel<<<1, 256, 0, stream>>>(weights, vocab, ntp, tids, tw);

    // 6) expansion generator MLP + pack outputs
    expand_kernel<<<1, 320, 0, stream>>>(
        emb, qctx, tids, tw, ntp, eW1, eb1, eW2, eb2, (float*)d_out);
}
